// CrossAttentionBlock_65309272703501
// MI455X (gfx1250) — matrix operand order repaired and checked by reading, not compiled, not run on hardware
//
#include <hip/hip_runtime.h>

// ---------------------------------------------------------------------------
// CrossAttentionBlock on MI455X (gfx1250), flash-attention formulation, v3b.
//
// B=4, C_hsi=128, C_msi=64, O=128, N=4096.
// - 256-thread blocks (8 wave32), each wave owns a 16-row query tile.
// - K/V staged in LDS per 64-column step via GLOBAL_LOAD_ASYNC_TO_LDS_B128
//   (ASYNCcnt) when available, else reg-copy fallback.
// - WMMA B-tile ds_loads software-pipelined (double-buffered) so the matrix
//   pipe is not serialized on s_wait_dscnt 0x0.
// - Online softmax: max via width-16 shuffles, row-sum via ones-matrix WMMA.
// - 34 v_wmma_f32_16x16x32_bf16 per 64-j step per wave.
// ---------------------------------------------------------------------------

typedef __bf16 bf16;
typedef __attribute__((ext_vector_type(16))) __bf16 v16bf;
typedef __attribute__((ext_vector_type(8)))  __bf16 v8bf;
typedef __attribute__((ext_vector_type(4)))  __bf16 v4bf;
typedef __attribute__((ext_vector_type(8)))  float  v8f;
typedef __attribute__((ext_vector_type(4)))  float  v4f;

#define BATCH 4
#define O_DIM 128
#define N_PIX 4096
#define C_HSI 128
#define C_MSI 64
#define JT    64          // j-columns per streaming step

#if __has_builtin(__builtin_amdgcn_global_load_async_to_lds_b128)
#define HAVE_ASYNC_LDS 1
#else
#define HAVE_ASYNC_LDS 0
#endif

#if HAVE_ASYNC_LDS
// Builtin signature (from hipcc diagnostic): param0 = v4i* in AS(1) (non-const),
// param1 = v4i* in AS(3); offset/cpol immediates.
typedef int v4i __attribute__((vector_size(16)));
typedef __attribute__((address_space(1))) v4i gv4i;
typedef __attribute__((address_space(3))) v4i lv4i;

static __device__ inline void async_ld_b128(const bf16* g, bf16* l) {
  __builtin_amdgcn_global_load_async_to_lds_b128(
      (gv4i*)(v4i*)(size_t)g,      // const/generic -> AS1 v4i*
      (lv4i*)(v4i*)l,              // generic LDS  -> AS3 v4i*
      0, 0);
}
#endif

static __device__ inline void wait_asynccnt0() {
#if __has_builtin(__builtin_amdgcn_s_wait_asynccnt)
  __builtin_amdgcn_s_wait_asynccnt(0);
#else
  asm volatile("s_wait_asynccnt 0x0" ::: "memory");
#endif
}

// ---------------------------------------------------------------------------
// Projection storing [b][n][O] (contiguous in o) — feeds WMMA A (Q) and the
// first GEMM's B (K) tiles as contiguous 16B/32B loads.
// ---------------------------------------------------------------------------
__global__ __launch_bounds__(O_DIM) void proj_nO(const float* __restrict__ z,
    const float* __restrict__ W, const float* __restrict__ bias,
    bf16* __restrict__ dst, int C)
{
  const int o = threadIdx.x;
  const int n = blockIdx.x;
  const int b = blockIdx.y;
  const float* zp = z + (size_t)b * C * N_PIX + n;   // uniform across block -> s_loads
  const float* wp = W + (size_t)o * C;
  float acc = bias[o];
  for (int c = 0; c < C; ++c)
    acc = fmaf(wp[c], zp[(size_t)c * N_PIX], acc);
  dst[((size_t)b * N_PIX + n) * O_DIM + o] = (bf16)acc;  // coalesced
}

// ---------------------------------------------------------------------------
// Projection storing [b][O][N] (contiguous in n) — feeds the second GEMM's
// B (V) tiles as contiguous 32B loads.
// ---------------------------------------------------------------------------
__global__ __launch_bounds__(256) void proj_On(const float* __restrict__ z,
    const float* __restrict__ W, const float* __restrict__ bias,
    bf16* __restrict__ dst, int C)
{
  const int n = blockIdx.x * 256 + threadIdx.x;
  const int o = blockIdx.y;
  const int b = blockIdx.z;
  const float* zp = z + (size_t)b * C * N_PIX + n;   // coalesced
  const float* wp = W + (size_t)o * C;               // uniform -> s_loads
  float acc = bias[o];
  for (int c = 0; c < C; ++c)
    acc = fmaf(wp[c], zp[(size_t)c * N_PIX], acc);
  dst[((size_t)b * O_DIM + o) * N_PIX + n] = (bf16)acc;  // coalesced
}

static __device__ inline v16bf pack16(v8bf lo, v8bf hi8) {
  v16bf r;
#pragma unroll
  for (int e = 0; e < 8; ++e) { r[e] = lo[e]; r[e + 8] = hi8[e]; }
  return r;
}

// ---------------------------------------------------------------------------
// Flash attention: grid (N_PIX/128, BATCH), block 256 (8 wave32).
// ---------------------------------------------------------------------------
__global__ __launch_bounds__(256) void attn_kernel(
    const bf16* __restrict__ Qh, const bf16* __restrict__ Kh,
    const bf16* __restrict__ Vh, const float* __restrict__ z_hsi,
    const float* __restrict__ gamma, float* __restrict__ out)
{
  __shared__ bf16 kbuf[JT * O_DIM];      // K tile  [jl][o]   16 KB
  __shared__ bf16 vbuf[O_DIM * JT];      // V tile  [o][jl]   16 KB
  __shared__ bf16 pbuf[8][16 * JT];      // per-wave P tile   16 KB (2 KB each)

  const int wave = threadIdx.x >> 5;
  const int lane = threadIdx.x & 31;
  const int row  = lane & 15;            // A row / B,D column index
  const int hi   = lane >> 4;            // half-wave select
  const int b    = blockIdx.y;
  const int i0   = (blockIdx.x * 8 + wave) * 16;

  // Resident Q A-tiles (16 x 32 bf16, 4 chunks over o=128).
  v16bf qa[4];
  {
    const bf16* qb = Qh + ((size_t)(b * N_PIX + i0 + row)) * O_DIM + hi * 16;
#pragma unroll
    for (int c = 0; c < 4; ++c) {
      v8bf lo  = *(const v8bf*)(qb + c * 32);
      v8bf hi8 = *(const v8bf*)(qb + c * 32 + 8);
      qa[c] = pack16(lo, hi8);
    }
  }

  v16bf ones;                            // B-matrix of 1.0 for row-sum WMMA
#pragma unroll
  for (int e = 0; e < 16; ++e) ones[e] = (bf16)1.0f;

  float mrun[8], lrun[8], corrA[8];
  v8f acc[8];                            // out^T accumulators, 8 o-chunks
#pragma unroll
  for (int d = 0; d < 8; ++d) { mrun[d] = -INFINITY; lrun[d] = 0.0f; }
#pragma unroll
  for (int oc = 0; oc < 8; ++oc) { v8f z = {}; acc[oc] = z; }

  bf16* pw = pbuf[wave];

  for (int j0 = 0; j0 < N_PIX; j0 += JT) {
    // ---- cooperative LDS staging of K/V tiles --------------------------
    __syncthreads();                       // previous iteration done reading
    {
      // K rows j0..j0+63 contiguous in [b][n][O]: straight 16KB block copy.
      const bf16* ks = Kh + ((size_t)(b * N_PIX + j0)) * O_DIM;
      const bf16* vs = Vh + ((size_t)b * O_DIM) * N_PIX + j0;
#if HAVE_ASYNC_LDS
#pragma unroll
      for (int r = 0; r < 4; ++r) {
        int idx = threadIdx.x + r * 256;   // 1024 x b128 per 16KB buffer
        async_ld_b128(ks + (size_t)idx * 8, kbuf + idx * 8);
        int o = idx >> 3, part = idx & 7;  // 8 x b128 per 64-col V row
        async_ld_b128(vs + (size_t)o * N_PIX + part * 8, vbuf + o * JT + part * 8);
      }
      wait_asynccnt0();
#else
      v16bf* kd = (v16bf*)kbuf;            // 512 x v16bf
      v16bf* vd = (v16bf*)vbuf;            // 512 x v16bf
#pragma unroll
      for (int r = 0; r < 2; ++r) {
        int idx = threadIdx.x + r * 256;
        kd[idx] = ((const v16bf*)ks)[idx];
        int o = idx >> 2, part = idx & 3;  // 4 x v16bf per 64-col V row
        vd[idx] = *(const v16bf*)(vs + (size_t)o * N_PIX + part * 16);
      }
#endif
    }
    __syncthreads();

    // ---- S = Q^T K : 4 j-subtiles (subtile t = columns j0 + 4c + t) -----
    // K B-tile loads double-buffered across o-chunks c to keep ds loads in
    // flight while WMMAs execute.
    v8f s[4];
#pragma unroll
    for (int t = 0; t < 4; ++t) { v8f z = {}; s[t] = z; }
    v16bf kb[2][4];
#pragma unroll
    for (int t = 0; t < 4; ++t)
      kb[0][t] = *(const v16bf*)(kbuf + (4 * row + t) * O_DIM + hi * 16);
#pragma unroll
    for (int c = 0; c < 4; ++c) {
      if (c < 3) {
#pragma unroll
        for (int t = 0; t < 4; ++t)
          kb[(c + 1) & 1][t] =
              *(const v16bf*)(kbuf + (4 * row + t) * O_DIM + (c + 1) * 32 + hi * 16);
      }
#pragma unroll
      for (int t = 0; t < 4; ++t)
        s[t] = __builtin_amdgcn_wmma_f32_16x16x32_bf16(false, qa[c], false,
                                                       kb[c & 1][t],
                                                       (short)0, s[t], false, false);
    }

    // ---- online softmax: max via shuffles, P packed to LDS --------------
#pragma unroll
    for (int d = 0; d < 8; ++d) {
      float mx = fmaxf(fmaxf(s[0][d], s[1][d]), fmaxf(s[2][d], s[3][d]));
#pragma unroll
      for (int m = 1; m <= 8; m <<= 1) mx = fmaxf(mx, __shfl_xor(mx, m, 16));
      float mnew = fmaxf(mrun[d], mx);
      float corr = __expf(mrun[d] - mnew);
      corrA[d] = corr;
      mrun[d] = mnew;
      v4bf pk;
#pragma unroll
      for (int t = 0; t < 4; ++t) pk[t] = (bf16)__expf(s[t][d] - mnew);
      // row r = hi*8+d, local cols 4*row .. 4*row+3  (j_local = 4c+t)
      *(v4bf*)(pw + (hi * 8 + d) * JT + 4 * row) = pk;
#pragma unroll
      for (int oc = 0; oc < 8; ++oc) acc[oc][d] *= corr;   // rescale output
    }
    __builtin_amdgcn_wave_barrier();       // per-wave tile: DS is in-order

    // ---- acc += P V^T, and row-sum(P) via ones-matrix WMMA --------------
    // V B-tile loads double-buffered across the oc loop.
    v8f rs8 = {};
#pragma unroll
    for (int half = 0; half < 2; ++half) {
      const bf16* pl = pw + row * JT + half * 32 + hi * 16;
      v8bf plo = *(const v8bf*)pl;
      v8bf phi = *(const v8bf*)(pl + 8);
      v16bf pa = pack16(plo, phi);
      v16bf bv[2];
      bv[0] = *(const v16bf*)(vbuf + row * JT + half * 32 + hi * 16);
      rs8 = __builtin_amdgcn_wmma_f32_16x16x32_bf16(false, pa, false, ones,
                                                    (short)0, rs8, false, false);
#pragma unroll
      for (int oc = 0; oc < 8; ++oc) {
        if (oc < 7)
          bv[(oc + 1) & 1] = *(const v16bf*)(vbuf + ((oc + 1) * 16 + row) * JT +
                                             half * 32 + hi * 16);
        acc[oc] = __builtin_amdgcn_wmma_f32_16x16x32_bf16(false, pa, false,
                                                          bv[oc & 1],
                                                          (short)0, acc[oc], false, false);
      }
    }
    __builtin_amdgcn_wave_barrier();
#pragma unroll
    for (int d = 0; d < 8; ++d)
      lrun[d] = lrun[d] * corrA[d] + rs8[d];
  }

  // ---- epilogue: out = gamma * acc/l + z_hsi, coalesced via LDS transpose
  const float g = gamma[0];
  float rinv[8];
#pragma unroll
  for (int d = 0; d < 8; ++d) rinv[d] = 1.0f / lrun[d];

  float* scr = (float*)pbuf[wave];         // 16 x 17 padded f32 scratch (1088B)
#pragma unroll
  for (int oc = 0; oc < 8; ++oc) {
    __builtin_amdgcn_wave_barrier();
#pragma unroll
    for (int d = 0; d < 8; ++d)
      scr[(hi * 8 + d) * 17 + row] = acc[oc][d] * rinv[d];
    __builtin_amdgcn_wave_barrier();
#pragma unroll
    for (int rep = 0; rep < 2; ++rep) {
      int q  = lane + rep * 32;            // 64 quads: 16 o-cols x 4 row-quads
      int c  = q >> 2;                     // o-col within chunk
      int r0 = (q & 3) * 4;                // first of 4 consecutive i rows
      v4f v;
#pragma unroll
      for (int k = 0; k < 4; ++k) v[k] = scr[(r0 + k) * 17 + c];
      const size_t idx = ((size_t)b * O_DIM + oc * 16 + c) * N_PIX + i0 + r0;
      v4f zz = *(const v4f*)(z_hsi + idx);
      v4f o4;
#pragma unroll
      for (int k = 0; k < 4; ++k) o4[k] = g * v[k] + zz[k];
      *(v4f*)(out + idx) = o4;             // b128, 64B-contiguous per 4 lanes
    }
  }
}

// ---------------------------------------------------------------------------
extern "C" void kernel_launch(void* const* d_in, const int* in_sizes, int n_in,
                              void* d_out, int out_size, void* d_ws, size_t ws_size,
                              hipStream_t stream) {
  const float* z_hsi = (const float*)d_in[0];
  const float* z_msi = (const float*)d_in[1];
  const float* Wq    = (const float*)d_in[2];
  const float* bq    = (const float*)d_in[3];
  const float* Wk    = (const float*)d_in[4];
  const float* bk    = (const float*)d_in[5];
  const float* Wv    = (const float*)d_in[6];
  const float* bv    = (const float*)d_in[7];
  const float* gamma = (const float*)d_in[8];

  const size_t projElems = (size_t)BATCH * N_PIX * O_DIM;  // 2M bf16 each
  bf16* qh = (bf16*)d_ws;             // [b][n][O]
  bf16* kh = qh + projElems;          // [b][n][O]
  bf16* vh = kh + projElems;          // [b][O][N]

  proj_nO<<<dim3(N_PIX, BATCH), O_DIM, 0, stream>>>(z_hsi, Wq, bq, qh, C_HSI);
  proj_nO<<<dim3(N_PIX, BATCH), O_DIM, 0, stream>>>(z_msi, Wk, bk, kh, C_MSI);
  proj_On<<<dim3(N_PIX / 256, O_DIM, BATCH), 256, 0, stream>>>(z_msi, Wv, bv, vh, C_MSI);

  attn_kernel<<<dim3(N_PIX / 128, BATCH), 256, 0, stream>>>(
      qh, kh, vh, z_hsi, gamma, (float*)d_out);
}
